// TimeMixingTriton_24876450579372
// MI455X (gfx1250) — matrix operand order repaired and checked by reading, not compiled, not run on hardware
//
#include <hip/hip_runtime.h>
#include <hip/hip_bf16.h>
#include <math.h>

// ---------------- constants (reference shapes) ----------------
#define BB 2
#define TT 2048
#define CC 1024
#define HH 16
#define NN 64
#define MM (BB * TT)   // 4096 rows

typedef __attribute__((ext_vector_type(16))) __bf16 v16bf;
typedef __attribute__((ext_vector_type(8)))  __bf16 v8bf;
typedef __attribute__((ext_vector_type(8)))  float  v8f;

#define GAS __attribute__((address_space(1)))
#define LAS __attribute__((address_space(3)))

// ---- CDNA5 async-to-LDS path (guarded; falls back to register pipeline) ----
#if defined(__has_builtin)
#  if __has_builtin(__builtin_amdgcn_global_load_async_to_lds_b32)
#    define USE_ASYNC_LDS 1
#  else
#    define USE_ASYNC_LDS 0
#  endif
#  if __has_builtin(__builtin_amdgcn_s_wait_asynccnt)
#    define WAIT_ASYNC() __builtin_amdgcn_s_wait_asynccnt(0)
#  else
#    define WAIT_ASYNC() asm volatile("s_wait_asynccnt 0x0" ::: "memory")
#  endif
#else
#  define USE_ASYNC_LDS 0
#  define WAIT_ASYNC()
#endif

__device__ __forceinline__ float sigmoidf_(float x) {
    return 1.0f / (1.0f + __expf(-x));
}

#if USE_ASYNC_LDS
// builtin signature (from hipcc diagnostics): (global int*, shared int*, imm offset, imm cpol).
// generic->AS casts: on amdgcn the low 32 bits of a generic LDS address ARE the
// LDS offset (flat aperture rule), and AS1 shares the generic representation.
__device__ __forceinline__ void async_g2l_b32(const float* g, float* l) {
    __builtin_amdgcn_global_load_async_to_lds_b32(
        (GAS int*)(unsigned long long)(const void*)g,
        (LAS int*)(unsigned)(unsigned long long)(void*)l,
        0, 0);
}
#endif

// ---------------- f32 -> bf16 conversion ----------------
__global__ __launch_bounds__(256) void cvt_bf16_kernel(const float* __restrict__ src,
                                                       __bf16* __restrict__ dst, int n) {
    int i = blockIdx.x * blockDim.x + threadIdx.x;
    if (i < n) dst[i] = (__bf16)src[i];
}

// ---------------- token shift: 6 shifted operands in bf16 ----------------
__global__ __launch_bounds__(256) void shift_kernel(
    const float* __restrict__ x,
    const float* __restrict__ tr, const float* __restrict__ tw,
    const float* __restrict__ tk, const float* __restrict__ tv,
    const float* __restrict__ ta, const float* __restrict__ tg,
    __bf16* __restrict__ xr, __bf16* __restrict__ xw, __bf16* __restrict__ xk,
    __bf16* __restrict__ xv, __bf16* __restrict__ xa, __bf16* __restrict__ xg) {
    int idx = blockIdx.x * blockDim.x + threadIdx.x;
    if (idx >= MM * CC) return;
    int c = idx & (CC - 1);
    int t = (idx / CC) % TT;
    float xc = x[idx];
    float xp = (t == 0) ? 0.0f : x[idx - CC];
    float xx = xp - xc;
    xr[idx] = (__bf16)(xc + xx * tr[c]);
    xw[idx] = (__bf16)(xc + xx * tw[c]);
    xk[idx] = (__bf16)(xc + xx * tk[c]);
    xv[idx] = (__bf16)(xc + xx * tv[c]);
    xa[idx] = (__bf16)(xc + xx * ta[c]);
    xg[idx] = (__bf16)(xc + xx * tg[c]);
}

// ---------------- WMMA GEMM:  out[m,n] = sum_k A[m,k] * W[n,k]  ----------------
// A: (M,K) bf16 row-major.  W: (N,K) bf16 row-major (i.e. we compute A @ W^T).
// Block = 128 threads = 4 waves; each wave owns a 16(M) x 64(N) strip.
enum { EPI_F32 = 0, EPI_BF16, EPI_TANH_BF16, EPI_SIG_BF16, EPI_W_F32, EPI_SIGB_F32 };

template <int EPI>
__global__ __launch_bounds__(128) void gemm_wmma_kernel(
    const __bf16* __restrict__ A, const __bf16* __restrict__ Wt,
    const float* __restrict__ bias, float* __restrict__ outF,
    __bf16* __restrict__ outB, int Mtiles /* M/64 */, int K, int Nsz) {
    const int wave = threadIdx.x >> 5;
    const int lane = threadIdx.x & 31;
    const int half = lane >> 4;     // 0: lanes 0-15, 1: lanes 16-31
    const int l16  = lane & 15;

    const int bm = blockIdx.x % Mtiles;
    const int bn = blockIdx.x / Mtiles;
    const int m0 = bm * 64 + wave * 16;
    const int n0 = bn * 64;

    v8f acc0 = {}, acc1 = {}, acc2 = {}, acc3 = {};

    const __bf16* __restrict__ Arow = A + (size_t)(m0 + l16) * K;
    const __bf16* __restrict__ Wrow = Wt + (size_t)(n0 + l16) * K;
    const size_t wstride = (size_t)16 * K;  // 16 N rows per subtile

    for (int kb = 0; kb < K; kb += 32) {
        // A fragment: lane holds M=m0+l16, K = kb + half*16 + [0..15] (contiguous)
        v8bf alo = *(const v8bf*)(Arow + kb + half * 16);
        v8bf ahi = *(const v8bf*)(Arow + kb + half * 16 + 8);
        v16bf a;
#pragma unroll
        for (int i = 0; i < 8; ++i) { a[i] = alo[i]; a[i + 8] = ahi[i]; }

        // B fragments: lane holds N=n0+j*16+l16, K = kb + half*16 + [0..15] (contiguous)
        v16bf b0 = *(const v16bf*)(Wrow + 0 * wstride + kb + half * 16);
        v16bf b1 = *(const v16bf*)(Wrow + 1 * wstride + kb + half * 16);
        v16bf b2 = *(const v16bf*)(Wrow + 2 * wstride + kb + half * 16);
        v16bf b3 = *(const v16bf*)(Wrow + 3 * wstride + kb + half * 16);

        if (kb + 32 < K) {  // prefetch next K slab into near cache (locality 3 -> WGP scope)
            __builtin_prefetch(Arow + kb + 32, 0, 3);
            __builtin_prefetch(Wrow + kb + 32, 0, 3);
        }

        acc0 = __builtin_amdgcn_wmma_f32_16x16x32_bf16(false, a, false, b0, (short)0, acc0, false, false);
        acc1 = __builtin_amdgcn_wmma_f32_16x16x32_bf16(false, a, false, b1, (short)0, acc1, false, false);
        acc2 = __builtin_amdgcn_wmma_f32_16x16x32_bf16(false, a, false, b2, (short)0, acc2, false, false);
        acc3 = __builtin_amdgcn_wmma_f32_16x16x32_bf16(false, a, false, b3, (short)0, acc3, false, false);
    }

    // C/D layout: VGPR i, lanes 0-15 -> M = m0 + i, lanes 16-31 -> M = m0 + 8 + i; N = n0 + j*16 + l16
    const int mbase = m0 + half * 8;
    auto store_tile = [&](const v8f& acc, int j) {
        const int n = n0 + j * 16 + l16;
#pragma unroll
        for (int i = 0; i < 8; ++i) {
            float v = acc[i];
            size_t o = (size_t)(mbase + i) * Nsz + n;
            if constexpr (EPI == EPI_F32) {
                outF[o] = v;
            } else if constexpr (EPI == EPI_BF16) {
                outB[o] = (__bf16)v;
            } else if constexpr (EPI == EPI_TANH_BF16) {
                outB[o] = (__bf16)tanhf(v);
            } else if constexpr (EPI == EPI_SIG_BF16) {
                outB[o] = (__bf16)sigmoidf_(v);
            } else if constexpr (EPI == EPI_W_F32) {
                outF[o] = __expf(-0.606531f * sigmoidf_(bias[n] + v));
            } else {  // EPI_SIGB_F32
                outF[o] = sigmoidf_(bias[n] + v);
            }
        }
    };
    store_tile(acc0, 0);
    store_tile(acc1, 1);
    store_tile(acc2, 2);
    store_tile(acc3, 3);
}

// ---------------- k/v mixing ----------------
__global__ __launch_bounds__(256) void mix_kv_kernel(
    float* __restrict__ k, float* __restrict__ v,
    const float* __restrict__ a, const float* __restrict__ vmix,
    const float* __restrict__ v_first, const float* __restrict__ k_a) {
    int idx = blockIdx.x * blockDim.x + threadIdx.x;
    if (idx >= MM * CC) return;
    int c = idx & (CC - 1);
    k[idx] = k[idx] * (1.0f + (a[idx] - 1.0f) * k_a[c]);
    float vv = v[idx];
    v[idx] = vv + (v_first[idx] - vv) * vmix[idx];
}

// ---------------- WKV7 sequential scan (software-pipelined) ----------------
// one block per (b,h); 64 threads; thread i owns state row i (64 f32 in VGPRs)
// state[i][j] = state[i][j]*w[i] + k[i]*v[j];  y[i] = sum_j state[i][j]*r[j]
__global__ __launch_bounds__(64) void wkv7_scan_kernel(
    const float* __restrict__ r, const float* __restrict__ w,
    const float* __restrict__ k, const float* __restrict__ v,
    float* __restrict__ y) {
    const int bh = blockIdx.x;           // 0..B*H-1
    const int b = bh / HH, h = bh % HH;
    const int i = threadIdx.x;           // 0..63

    __shared__ float rs[2][NN];
    __shared__ float vs[2][NN];

    float st[NN];
#pragma unroll
    for (int j = 0; j < NN; ++j) st[j] = 0.0f;

    const size_t base = (size_t)b * TT * CC + (size_t)h * NN + i;
    const float* rp = r + base;
    const float* wp = w + base;
    const float* kp = k + base;
    const float* vp = v + base;
    float* yp = y + base;

#if USE_ASYNC_LDS
    // t=0 broadcasts go memory->LDS with no VGPR round-trip (ASYNCcnt-tracked)
    async_g2l_b32(rp, &rs[0][i]);
    async_g2l_b32(vp, &vs[0][i]);
    float w_n = wp[0];
    float k_n = kp[0];
    WAIT_ASYNC();
    __syncthreads();
    for (int t = 0; t < TT; ++t) {
        const int buf = t & 1;
        const float wi = w_n, ki = k_n;
        if (t + 1 < TT) {  // overlap next step's loads with this step's FMA loop
            const size_t o = (size_t)(t + 1) * CC;
            async_g2l_b32(rp + o, &rs[buf ^ 1][i]);
            async_g2l_b32(vp + o, &vs[buf ^ 1][i]);
            w_n = wp[o];
            k_n = kp[o];
        }
        float acc = 0.0f;
#pragma unroll
        for (int j = 0; j < NN; ++j) {
            st[j] = st[j] * wi + ki * vs[buf][j];
            acc += st[j] * rs[buf][j];
        }
        yp[(size_t)t * CC] = acc;
        WAIT_ASYNC();
        __syncthreads();
    }
#else
    // register-pipelined fallback
    float r_n = rp[0], v_n = vp[0], w_n = wp[0], k_n = kp[0];
    for (int t = 0; t < TT; ++t) {
        const int buf = t & 1;
        rs[buf][i] = r_n;
        vs[buf][i] = v_n;
        const float wi = w_n, ki = k_n;
        __syncthreads();
        if (t + 1 < TT) {  // loads overlap the FMA loop below
            const size_t o = (size_t)(t + 1) * CC;
            r_n = rp[o]; v_n = vp[o]; w_n = wp[o]; k_n = kp[o];
        }
        float acc = 0.0f;
#pragma unroll
        for (int j = 0; j < NN; ++j) {
            st[j] = st[j] * wi + ki * vs[buf][j];
            acc += st[j] * rs[buf][j];
        }
        yp[(size_t)t * CC] = acc;
    }
#endif
}

// ---------------- LayerNorm + rkv + gating -> bf16 operand for final GEMM ----------------
__global__ __launch_bounds__(256) void ln_rkv_g_kernel(
    const float* __restrict__ y, const float* __restrict__ r,
    const float* __restrict__ k, const float* __restrict__ v,
    const float* __restrict__ g, const float* __restrict__ r_k,
    const float* __restrict__ ln_g, const float* __restrict__ ln_b,
    __bf16* __restrict__ yg) {
    const int row = blockIdx.x;               // 0..M-1
    const int tid = threadIdx.x;              // 256 threads
    const size_t ro = (size_t)row * CC;

    __shared__ float s1[256], s2[256];
    __shared__ float hsum[HH];

    float vals[4];
    float s = 0.0f, sq = 0.0f;
#pragma unroll
    for (int q = 0; q < 4; ++q) {
        float yv = y[ro + tid + q * 256];
        vals[q] = yv;
        s += yv;
        sq += yv * yv;
    }
    s1[tid] = s;
    s2[tid] = sq;
    if (tid < HH) hsum[tid] = 0.0f;
    __syncthreads();
    for (int stp = 128; stp > 0; stp >>= 1) {
        if (tid < stp) { s1[tid] += s1[tid + stp]; s2[tid] += s2[tid + stp]; }
        __syncthreads();
    }
    const float mean = s1[0] * (1.0f / CC);
    const float var  = s2[0] * (1.0f / CC) - mean * mean;
    const float rstd = rsqrtf(var + 1e-5f);

    // per-head reduction: sum_n r*k*r_k  (ds_add_f32 atomics into 16 slots)
#pragma unroll
    for (int q = 0; q < 4; ++q) {
        int c = tid + q * 256;
        float p = r[ro + c] * k[ro + c] * r_k[c];
        atomicAdd(&hsum[c >> 6], p);
    }
    __syncthreads();

#pragma unroll
    for (int q = 0; q < 4; ++q) {
        int c = tid + q * 256;
        float ln = (vals[q] - mean) * rstd * ln_g[c] + ln_b[c];
        float rkv = hsum[c >> 6] * v[ro + c];
        yg[ro + c] = (__bf16)((ln + rkv) * g[ro + c]);
    }
}

// ---------------- host side ----------------
extern "C" void kernel_launch(void* const* d_in, const int* in_sizes, int n_in,
                              void* d_out, int out_size, void* d_ws, size_t ws_size,
                              hipStream_t stream) {
    const float* x       = (const float*)d_in[0];
    const float* v_first = (const float*)d_in[1];
    // weights d_in[2..13]: Wr Wk Wv Wo Wg1 Wg2 W1 W2 A1 A2 V1 V2
    const float* tm_r = (const float*)d_in[14];
    const float* tm_w = (const float*)d_in[15];
    const float* tm_k = (const float*)d_in[16];
    const float* tm_v = (const float*)d_in[17];
    const float* tm_a = (const float*)d_in[18];
    const float* tm_g = (const float*)d_in[19];
    const float* w0   = (const float*)d_in[20];
    const float* a0   = (const float*)d_in[21];
    const float* v0   = (const float*)d_in[22];
    const float* k_a  = (const float*)d_in[24];
    const float* r_k  = (const float*)d_in[25];
    const float* ln_g = (const float*)d_in[26];
    const float* ln_b = (const float*)d_in[27];
    float* out = (float*)d_out;

    // workspace carve-up
    size_t off = 0;
    auto alloc = [&](size_t bytes) -> void* {
        void* p = (char*)d_ws + off;
        off += (bytes + 255) & ~(size_t)255;
        return p;
    };
    const size_t actB = (size_t)MM * CC * sizeof(__bf16);
    const size_t actF = (size_t)MM * CC * sizeof(float);
    const size_t wB   = (size_t)CC * CC * sizeof(__bf16);

    __bf16* xb[6];  for (int i = 0; i < 6; ++i)  xb[i] = (__bf16*)alloc(actB);
    __bf16* wb[12]; for (int i = 0; i < 12; ++i) wb[i] = (__bf16*)alloc(wB);
    float* fr    = (float*)alloc(actF);
    float* fw    = (float*)alloc(actF);
    float* fk    = (float*)alloc(actF);
    float* fv    = (float*)alloc(actF);
    float* fa    = (float*)alloc(actF);
    float* fg    = (float*)alloc(actF);
    float* fvmix = (float*)alloc(actF);
    float* fy    = (float*)alloc(actF);
    __bf16* hw = (__bf16*)alloc(actB);
    __bf16* ha = (__bf16*)alloc(actB);
    __bf16* hg = (__bf16*)alloc(actB);
    __bf16* hv = (__bf16*)alloc(actB);
    __bf16* yg = (__bf16*)alloc(actB);

    // 1) convert all 12 weights to bf16
    for (int i = 0; i < 12; ++i)
        cvt_bf16_kernel<<<(CC * CC) / 256, 256, 0, stream>>>((const float*)d_in[2 + i], wb[i], CC * CC);

    // 2) token shift -> 6 bf16 GEMM operands (xr,xw,xk,xv,xa,xg)
    shift_kernel<<<(MM * CC) / 256, 256, 0, stream>>>(
        x, tm_r, tm_w, tm_k, tm_v, tm_a, tm_g,
        xb[0], xb[1], xb[2], xb[3], xb[4], xb[5]);

    // 3) projection GEMMs (M=4096, N=K=1024); grid = (M/64)*(N/64) blocks of 128 thr
    const int Mtiles = MM / 64;
    const int grid = Mtiles * (CC / 64);
    // wb: 0=Wr 1=Wk 2=Wv 3=Wo 4=Wg1 5=Wg2 6=W1 7=W2 8=A1 9=A2 10=V1 11=V2
    gemm_wmma_kernel<EPI_F32>      <<<grid, 128, 0, stream>>>(xb[0], wb[0],  nullptr, fr,    nullptr, Mtiles, CC, CC); // r
    gemm_wmma_kernel<EPI_F32>      <<<grid, 128, 0, stream>>>(xb[2], wb[1],  nullptr, fk,    nullptr, Mtiles, CC, CC); // k
    gemm_wmma_kernel<EPI_F32>      <<<grid, 128, 0, stream>>>(xb[3], wb[2],  nullptr, fv,    nullptr, Mtiles, CC, CC); // v
    gemm_wmma_kernel<EPI_TANH_BF16><<<grid, 128, 0, stream>>>(xb[1], wb[6],  nullptr, nullptr, hw,    Mtiles, CC, CC); // tanh(xw@W1^T)
    gemm_wmma_kernel<EPI_W_F32>    <<<grid, 128, 0, stream>>>(hw,    wb[7],  w0,      fw,    nullptr, Mtiles, CC, CC); // w decay
    gemm_wmma_kernel<EPI_BF16>     <<<grid, 128, 0, stream>>>(xb[4], wb[8],  nullptr, nullptr, ha,    Mtiles, CC, CC); // xa@A1^T
    gemm_wmma_kernel<EPI_SIGB_F32> <<<grid, 128, 0, stream>>>(ha,    wb[9],  a0,      fa,    nullptr, Mtiles, CC, CC); // a
    gemm_wmma_kernel<EPI_SIG_BF16> <<<grid, 128, 0, stream>>>(xb[5], wb[4],  nullptr, nullptr, hg,    Mtiles, CC, CC); // sig(xg@Wg1^T)
    gemm_wmma_kernel<EPI_F32>      <<<grid, 128, 0, stream>>>(hg,    wb[5],  nullptr, fg,    nullptr, Mtiles, CC, CC); // g
    gemm_wmma_kernel<EPI_BF16>     <<<grid, 128, 0, stream>>>(xb[3], wb[10], nullptr, nullptr, hv,    Mtiles, CC, CC); // xv@V1^T
    gemm_wmma_kernel<EPI_SIGB_F32> <<<grid, 128, 0, stream>>>(hv,    wb[11], v0,      fvmix, nullptr, Mtiles, CC, CC); // v-mix gate

    // 4) k/v mixing
    mix_kv_kernel<<<(MM * CC) / 256, 256, 0, stream>>>(fk, fv, fa, fvmix, v_first, k_a);

    // 5) sequential WKV7 scan: 32 (b,h) blocks, 64 threads each
    wkv7_scan_kernel<<<BB * HH, NN, 0, stream>>>(fr, fw, fk, fv, fy);

    // 6) LayerNorm + rkv + gate -> bf16 operand
    ln_rkv_g_kernel<<<MM, 256, 0, stream>>>(fy, fr, fk, fv, fg, r_k, ln_g, ln_b, yg);

    // 7) output projection: (y*g) @ Wo^T -> d_out (f32)
    gemm_wmma_kernel<EPI_F32><<<grid, 128, 0, stream>>>(yg, wb[3], nullptr, out, nullptr, Mtiles, CC, CC);

    (void)in_sizes; (void)n_in; (void)out_size; (void)ws_size;
}
